// Encoder_27470610825904
// MI455X (gfx1250) — matrix operand order repaired and checked by reading, not compiled, not run on hardware
//
#include <hip/hip_runtime.h>
#include <hip/hip_bf16.h>
#include <cstdint>

// ---------------- problem constants ----------------
#define CB   8            // batch
#define TT   10           // time steps
#define HH   48
#define WW   48
#define HWSZ (HH*WW)      // 2304
#define FF   64           // hidden features
#define NG   256          // 4*FF gate channels
#define MROWS (CB*HWSZ)   // 18432 spatial rows per step
#define OUTSZ (CB*HWSZ*FF)

typedef __attribute__((ext_vector_type(16))) __bf16 v16bf;
typedef __attribute__((ext_vector_type(8)))  float  v8f;

static __device__ __forceinline__ unsigned short f2bf(float x) {
    unsigned u = __float_as_uint(x);
    unsigned r = u + 0x7FFFu + ((u >> 16) & 1u);   // round-to-nearest-even
    return (unsigned short)(r >> 16);
}
static __device__ __forceinline__ float bf2f(unsigned short h) {
    return __uint_as_float(((unsigned)h) << 16);
}
static __device__ __forceinline__ float hsig(float x) {
    return fminf(fmaxf(0.2f * x + 0.5f, 0.f), 1.f);
}

// ---------------- utility kernels ----------------
__global__ void k_zero_u32(unsigned* __restrict__ p, int n) {
    int i = blockIdx.x * blockDim.x + threadIdx.x;
    if (i < n) p[i] = 0u;
}

__global__ void k_f32_to_bf16(const float* __restrict__ src,
                              unsigned short* __restrict__ dst, int n) {
    int i = blockIdx.x * blockDim.x + threadIdx.x;
    if (i < n) dst[i] = f2bf(src[i]);
}

// Pack [Wx ; Wh] into bf16, transposed to [256][Kpad] (K contiguous per row).
__global__ void k_pack_w(const float* __restrict__ Wx, const float* __restrict__ Wh,
                         unsigned short* __restrict__ out,
                         int Cin, int K1, int Ktot, int Kpad) {
    int i = blockIdx.x * blockDim.x + threadIdx.x;
    if (i >= NG * Kpad) return;
    int n  = i / Kpad;
    int kk = i - n * Kpad;
    float v = 0.f;
    if (kk < K1) {
        int kh = kk / (3 * Cin); int r = kk - kh * (3 * Cin);
        int kw = r / Cin;        int ci = r - kw * Cin;
        v = Wx[((kh * 3 + kw) * Cin + ci) * NG + n];
    } else if (kk < Ktot) {
        int k2 = kk - K1;
        int kh = k2 / (3 * FF);  int r = k2 - kh * (3 * FF);
        int kw = r / FF;         int ci = r - kw * FF;
        v = Wh[((kh * 3 + kw) * FF + ci) * NG + n];
    }
    out[(size_t)n * Kpad + kk] = f2bf(v);
}

// ---- implicit-GEMM A gather: one K pair-column x 4 M-rows, branchless ----
template<int CIN, int K1V, int KTOT>
static __device__ __forceinline__ void gatherA(
    const unsigned short* __restrict__ X, int xbStride,
    const unsigned short* __restrict__ Hp,
    int kc, int pi, const int an[4], const int ah[4], const int aw[4],
    unsigned aval[4])
{
    const int kk = kc + pi * 2;
    const bool isX = kk < K1V;
    int khx = kk / (3 * CIN); int rx = kk - khx * (3 * CIN);
    int kwx = rx / CIN;       int cix = rx - kwx * CIN;
    int k2  = kk - K1V;
    int khh = k2 / (3 * FF);  int rh = k2 - khh * (3 * FF);
    int kwh = rh / FF;        int cih = rh - kwh * FF;
    const int kh = isX ? khx : khh;
    const int kw = isX ? kwx : kwh;
    const int ci = isX ? cix : cih;
    const int cs = isX ? CIN : FF;
    const int bs = isX ? xbStride : (HWSZ * FF);
    const unsigned short* base = isX ? X : Hp;
    const bool kvalid = kk < KTOT;
    #pragma unroll
    for (int j = 0; j < 4; ++j) {
        int hh  = ah[j] + kh - 1;
        int wxx = aw[j] + kw - 1;
        bool ok = kvalid & ((unsigned)hh < HH) & ((unsigned)wxx < WW);
        size_t off = ok ? ((size_t)an[j] * bs + (hh * WW + wxx) * cs + ci)
                        : (size_t)0;
        unsigned v = *(const unsigned*)(base + off);   // unconditional load
        aval[j] = ok ? v : 0u;
    }
}

// =====================================================================
// Fused ConvLSTM step: implicit GEMM (bf16 WMMA, fp32 acc) + LSTM update
//   gates[M=18432][256] = conv3x3(X,Wx) + conv3x3(Hprev,Wh)
// One WG: 64 M-rows x all 256 gate channels (8 waves; each wave:
// 2 M-frags x 4 gate-frags = 8 accs, 8 WMMAs per 32-deep K chunk).
// Double-buffered LDS, single barrier per chunk: chunk k+1's
// global->LDS staging overlaps chunk k's WMMAs; no regs live across
// barriers (avoids scratch spills).
// =====================================================================
template<int CIN, int K1V, int KTOT, int KPAD, bool WSEQ>
__global__ __launch_bounds__(256)
__attribute__((amdgpu_waves_per_eu(2)))
void k_convlstm_step(
    const unsigned short* __restrict__ X, int xbStride,
    const unsigned short* __restrict__ Hp,
    const unsigned short* __restrict__ Wt,
    const float* __restrict__ bias,
    float* __restrict__ cbuf,
    unsigned short* __restrict__ hnext,
    float* __restrict__ hOut,            // non-null only on final step
    float* __restrict__ cOut,            // non-null only on final step
    unsigned short* __restrict__ seqSlot, int seqBatchStride)
{
    __shared__ __align__(16) unsigned At[2][64][20];   // 64 M-rows x 32 bf16
    __shared__ __align__(16) unsigned Bt[2][NG][20];   // 256 N-rows x 32 bf16

    const int tid   = threadIdx.x;
    const int mTile = blockIdx.x * 64;
    const int wave  = tid >> 5, lane = tid & 31;
    const int wm    = (wave & 1) * 32;     // wave's M offset (2 frags of 16)
    const int cg    = (wave >> 1) * 16;    // wave's channel group [cg, cg+16)
    const int khalf = lane >> 4;
    const int l15   = lane & 15;

    // ---- A staging geometry (hoisted out of the K loop) ----
    const int pi   = tid & 15;
    const int rowG = tid >> 4;
    int an[4], ah[4], aw[4];
    #pragma unroll
    for (int j = 0; j < 4; ++j) {
        int m = mTile + rowG * 4 + j;
        int n = m / HWSZ;
        int s = m - n * HWSZ;
        an[j] = n;
        ah[j] = s / WW;
        aw[j] = s - ah[j] * WW;
    }

    v8f acc[2][4];
    #pragma unroll
    for (int mf = 0; mf < 2; ++mf)
        #pragma unroll
        for (int g = 0; g < 4; ++g)
            acc[mf][g] = (v8f){};

    // ---- stage chunk 0 into buffer 0 (global -> regs -> LDS) ----
    {
        unsigned aval[4];
        gatherA<CIN, K1V, KTOT>(X, xbStride, Hp, 0, pi, an, ah, aw, aval);
        #pragma unroll
        for (int j = 0; j < 4; ++j)
            At[0][rowG * 4 + j][pi] = aval[j];
        #pragma unroll
        for (int i = 0; i < 4; ++i) {
            int row = (tid >> 2) + i * 64;
            int c4  = tid & 3;
            uint4 v = *(const uint4*)(Wt + (size_t)row * KPAD + c4 * 8);
            *(uint4*)&Bt[0][row][c4 * 4] = v;
        }
    }

    #pragma unroll 1
    for (int kc = 0; kc < KPAD; kc += 32) {
        const int buf = (kc >> 5) & 1;
        // publishes chunk kc's tiles; retires reads of the other buffer
        __syncthreads();

        // ---- stage chunk kc+32 into the other buffer (overlaps WMMAs) ----
        if (kc + 32 < KPAD) {
            unsigned aval[4];
            gatherA<CIN, K1V, KTOT>(X, xbStride, Hp, kc + 32, pi, an, ah, aw, aval);
            #pragma unroll
            for (int j = 0; j < 4; ++j)
                At[buf ^ 1][rowG * 4 + j][pi] = aval[j];
            #pragma unroll
            for (int i = 0; i < 4; ++i) {
                int row = (tid >> 2) + i * 64;
                int c4  = tid & 3;
                uint4 v = *(const uint4*)(Wt + (size_t)row * KPAD + (kc + 32) + c4 * 8);
                *(uint4*)&Bt[buf ^ 1][row][c4 * 4] = v;
            }
        }

        // ---- fragments (ISA 7.12.2 16-bit layouts) + 8 WMMAs ----
        union { v16bf v; unsigned u[8]; } afrag[2], bfrag;
        #pragma unroll
        for (int mf = 0; mf < 2; ++mf) {
            int mrow = wm + mf * 16 + l15;
            #pragma unroll
            for (int r = 0; r < 8; ++r)
                afrag[mf].u[r] = At[buf][mrow][(r >> 2) * 8 + khalf * 4 + (r & 3)];
        }
        #pragma unroll
        for (int g = 0; g < 4; ++g) {
            int nc = g * 64 + cg + l15;
            #pragma unroll
            for (int r = 0; r < 8; ++r)
                bfrag.u[r] = Bt[buf][nc][(r >> 2) * 8 + khalf * 4 + (r & 3)];
            acc[0][g] = __builtin_amdgcn_wmma_f32_16x16x32_bf16(
                false, afrag[0].v, false, bfrag.v, (short)0, acc[0][g], false, false);
            acc[1][g] = __builtin_amdgcn_wmma_f32_16x16x32_bf16(
                false, afrag[1].v, false, bfrag.v, (short)0, acc[1][g], false, false);
        }
    }

    // ---- fused LSTM epilogue ----
    const int f  = cg + l15;
    const float bi  = bias[f];
    const float bff = bias[64 + f];
    const float bcc = bias[128 + f];
    const float boo = bias[192 + f];

    #pragma unroll
    for (int mf = 0; mf < 2; ++mf) {
        int mBase = mTile + wm + mf * 16 + khalf * 8;
        #pragma unroll
        for (int r = 0; r < 8; ++r) {
            int m = mBase + r;
            size_t idx = (size_t)m * FF + f;
            float gi = hsig(acc[mf][0][r] + bi);
            float gf = hsig(acc[mf][1][r] + bff);
            float gc =      acc[mf][2][r] + bcc;
            float go = hsig(acc[mf][3][r] + boo);
            float c  = gf * cbuf[idx] + gi * tanhf(gc);
            float h  = go * tanhf(c);
            cbuf[idx] = c;
            unsigned short hb = f2bf(h);
            hnext[idx] = hb;
            if (hOut) { hOut[idx] = h; cOut[idx] = c; }
            if (WSEQ) {
                int n = m / HWSZ;
                int s = m - n * HWSZ;
                seqSlot[(size_t)n * seqBatchStride + s * FF + f] = hb;
            }
        }
    }
}

// ---------------- BatchNorm (training stats over all but channel) -----------
__global__ void k_bn_stats(const unsigned short* __restrict__ seq,
                           float* __restrict__ stats, int n)
{
    __shared__ float ssum[64], ssq[64];
    int tid = threadIdx.x;
    if (tid < 64) { ssum[tid] = 0.f; ssq[tid] = 0.f; }
    __syncthreads();
    float s = 0.f, q = 0.f;
    int stride = gridDim.x * blockDim.x;      // multiple of 64 -> fixed channel
    for (int i = blockIdx.x * blockDim.x + tid; i < n; i += stride) {
        float v = bf2f(seq[i]);
        s += v; q += v * v;
    }
    atomicAdd(&ssum[tid & 63], s);
    atomicAdd(&ssq[tid & 63], q);
    __syncthreads();
    if (tid < 64) {
        atomicAdd(&stats[tid],      ssum[tid]);
        atomicAdd(&stats[64 + tid], ssq[tid]);
    }
}

__global__ void k_bn_finalize(float* __restrict__ stats,
                              const float* __restrict__ gamma,
                              const float* __restrict__ beta, float invN)
{
    int f = threadIdx.x;
    if (f >= 64) return;
    float mean = stats[f] * invN;
    float var  = stats[64 + f] * invN - mean * mean;
    float sc   = gamma[f] * rsqrtf(var + 1e-3f);
    stats[128 + f] = sc;
    stats[192 + f] = beta[f] - mean * sc;
}

__global__ void k_bn_apply(const unsigned short* __restrict__ seq,
                           unsigned short* __restrict__ out,
                           const float* __restrict__ stats, int n)
{
    int i = blockIdx.x * blockDim.x + threadIdx.x;
    if (i >= n) return;
    int f = i & 63;
    out[i] = f2bf(bf2f(seq[i]) * stats[128 + f] + stats[192 + f]);
}

// ---------------- host orchestration ----------------
extern "C" void kernel_launch(void* const* d_in, const int* in_sizes, int n_in,
                              void* d_out, int out_size, void* d_ws, size_t ws_size,
                              hipStream_t stream)
{
    const float* x      = (const float*)d_in[0];
    const float* Wx1    = (const float*)d_in[1];
    const float* Wh1    = (const float*)d_in[2];
    const float* b1     = (const float*)d_in[3];
    const float* gamma1 = (const float*)d_in[4];
    const float* beta1  = (const float*)d_in[5];
    const float* Wx2    = (const float*)d_in[6];
    const float* Wh2    = (const float*)d_in[7];
    const float* b2     = (const float*)d_in[8];
    float* out = (float*)d_out;

    char*  ws  = (char*)d_ws;
    size_t off = 0;
    auto carve = [&](size_t bytes) -> void* {
        void* p = ws + off;
        off = (off + bytes + 255) & ~(size_t)255;
        return p;
    };
    const int XN    = CB * TT * HWSZ * 16;        // 2,949,120
    const int KPAD1 = 736, K1_1 = 144, KT_1 = 720;
    const int KPAD2 = 1152, K1_2 = 576, KT_2 = 1152;
    const int SEQN  = CB * TT * HWSZ * FF;        // 11,796,480

    unsigned short* X16     = (unsigned short*)carve((size_t)XN * 2);
    unsigned short* W1t     = (unsigned short*)carve((size_t)NG * KPAD1 * 2);
    unsigned short* W2t     = (unsigned short*)carve((size_t)NG * KPAD2 * 2);
    unsigned short* seqRaw  = (unsigned short*)carve((size_t)SEQN * 2);
    unsigned short* seqNorm = (unsigned short*)carve((size_t)SEQN * 2);
    unsigned short* hA      = (unsigned short*)carve((size_t)MROWS * FF * 2);
    unsigned short* hB      = (unsigned short*)carve((size_t)MROWS * FF * 2);
    float*          cbuf    = (float*)carve((size_t)MROWS * FF * 4);
    float*          stats   = (float*)carve(256 * 4);
    (void)ws_size; (void)in_sizes; (void)n_in; (void)out_size;

    const int TPB = 256;
    auto blocks = [](int n, int t) { return (n + t - 1) / t; };

    // prep
    k_f32_to_bf16<<<blocks(XN, TPB), TPB, 0, stream>>>(x, X16, XN);
    k_pack_w<<<blocks(NG * KPAD1, TPB), TPB, 0, stream>>>(Wx1, Wh1, W1t, 16, K1_1, KT_1, KPAD1);
    k_pack_w<<<blocks(NG * KPAD2, TPB), TPB, 0, stream>>>(Wx2, Wh2, W2t, 64, K1_2, KT_2, KPAD2);
    k_zero_u32<<<1, 128, 0, stream>>>((unsigned*)stats, 128);

    const int stepGrid = MROWS / 64;              // 288 blocks
    const int ewN      = MROWS * FF;

    // ---------------- layer 1 ----------------
    k_zero_u32<<<blocks(ewN / 2, TPB), TPB, 0, stream>>>((unsigned*)hA, ewN / 2);
    k_zero_u32<<<blocks(ewN, TPB), TPB, 0, stream>>>((unsigned*)cbuf, ewN);
    {
        unsigned short* hc = hA;
        unsigned short* hn = hB;
        for (int t = 0; t < TT; ++t) {
            bool last = (t == TT - 1);
            k_convlstm_step<16, K1_1, KT_1, KPAD1, true><<<stepGrid, TPB, 0, stream>>>(
                X16 + (size_t)t * HWSZ * 16, TT * HWSZ * 16,
                hc, W1t, b1, cbuf, hn,
                last ? (out + 0) : nullptr,
                last ? (out + OUTSZ) : nullptr,
                seqRaw + (size_t)t * HWSZ * FF, TT * HWSZ * FF);
            unsigned short* tmp = hc; hc = hn; hn = tmp;
        }
    }

    // ---------------- batchnorm ----------------
    k_bn_stats<<<1440, TPB, 0, stream>>>(seqRaw, stats, SEQN);
    k_bn_finalize<<<1, 64, 0, stream>>>(stats, gamma1, beta1, 1.0f / (float)SEQN);
    k_bn_apply<<<blocks(SEQN, TPB), TPB, 0, stream>>>(seqRaw, seqNorm, stats, SEQN);

    // ---------------- layer 2 ----------------
    k_zero_u32<<<blocks(ewN / 2, TPB), TPB, 0, stream>>>((unsigned*)hA, ewN / 2);
    k_zero_u32<<<blocks(ewN, TPB), TPB, 0, stream>>>((unsigned*)cbuf, ewN);
    {
        unsigned short* hc = hA;
        unsigned short* hn = hB;
        for (int t = 0; t < TT; ++t) {
            bool last = (t == TT - 1);
            k_convlstm_step<64, K1_2, KT_2, KPAD2, false><<<stepGrid, TPB, 0, stream>>>(
                seqNorm + (size_t)t * HWSZ * FF, TT * HWSZ * FF,
                hc, W2t, b2, cbuf, hn,
                last ? (out + 2 * OUTSZ) : nullptr,
                last ? (out + 3 * OUTSZ) : nullptr,
                (unsigned short*)nullptr, 0);
            unsigned short* tmp = hc; hc = hn; hn = tmp;
        }
    }
}
